// SlotAttention_335007449589
// MI455X (gfx1250) — matrix operand order repaired and checked by reading, not compiled, not run on hardware
//
#include <hip/hip_runtime.h>
#include <hip/hip_bf16.h>

// ---------------------------------------------------------------------------
// CDNA5 (gfx1250) Slot Attention.  All GEMMs via v_wmma_f32_16x16x32_f16.
// k stored row-major f16, v stored transposed f16 (vT[d][j]) so every WMMA
// operand is assembled from 16/32-byte contiguous loads per lane.
// Input LN tiles are fetched with the Tensor Data Mover (tensor_load_to_lds,
// TENSORcnt) using the D# pad feature to produce the bank-staggered LDS
// layout directly.  Streaming GEMM loops are software-pipelined and use
// global_prefetch.
// ---------------------------------------------------------------------------

typedef __attribute__((ext_vector_type(16))) _Float16 v16h;
typedef __attribute__((ext_vector_type(8)))  _Float16 v8h;
typedef __attribute__((ext_vector_type(8)))  float    v8f;
typedef __attribute__((ext_vector_type(4)))  unsigned int u32x4;
typedef __attribute__((ext_vector_type(8)))  int      i32x8;
typedef __attribute__((ext_vector_type(4)))  int      i32x4;

#define DIMD   256
#define NTOK   4096
#define NB     32
#define NSLOTP 16   // slots padded 8 -> 16
#define HID    256

__device__ __forceinline__ v8f wmma32(v16h a, v16h b, v8f c) {
  // D = A(16x32 f16) * B(32x16 f16) + C(16x16 f32)
  return __builtin_amdgcn_wmma_f32_16x16x32_f16(false, a, false, b, (short)0, c,
                                                false, false);
}

// A-operand fragment: src is row-major [M x K]; lane l holds row (l&15),
// lanes 0-15: K = cb+[0,8) and cb+[16,24); lanes 16-31: cb+[8,16), cb+[24,32).
__device__ __forceinline__ v16h fragA(const _Float16* base, int row_base,
                                      int col_base, int ld) {
  const int lane = threadIdx.x & 31;
  const int l = lane & 15, hi = lane >> 4;
  const _Float16* p = base + (size_t)(row_base + l) * ld + col_base + hi * 8;
  v8h a = *(const v8h*)(p);
  v8h b = *(const v8h*)(p + 16);
  v16h r;
#pragma unroll
  for (int i = 0; i < 8; ++i) { r[i] = a[i]; r[i + 8] = b[i]; }
  return r;
}

// B-operand fragment: src is row-major [N x K] (i.e. B transposed in memory,
// e.g. weight[out, in] or k[token, d] or vT[d, j]).  Lane l holds column
// n=(l&15); lanes 0-15: K = cb+[0,8) and cb+[16,24); lanes 16-31: cb+[8,16), cb+[24,32).
__device__ __forceinline__ v16h fragB(const _Float16* base, int row_base,
                                      int col_base, int ld) {
  const int lane = threadIdx.x & 31;
  const int l = lane & 15, hi = lane >> 4;
  const _Float16* p = base + (size_t)(row_base + l) * ld + col_base + hi * 8;
  v8h a = *(const v8h*)(p);
  v8h b = *(const v8h*)(p + 16);
  v16h r;
#pragma unroll
  for (int i = 0; i < 8; ++i) { r[i] = a[i]; r[i + 8] = b[i]; }
  return r;
}

// C/D store: VGPR r -> row hi*8+r, col lane&15.
__device__ __forceinline__ void stC_f32(float* out, int row_base, int col_base,
                                        int ld, v8f c) {
  const int lane = threadIdx.x & 31;
  const int l = lane & 15, hi = lane >> 4;
#pragma unroll
  for (int r = 0; r < 8; ++r)
    out[(size_t)(row_base + hi * 8 + r) * ld + col_base + l] = c[r];
}
__device__ __forceinline__ void stC_f16(_Float16* out, int row_base,
                                        int col_base, int ld, v8f c) {
  const int lane = threadIdx.x & 31;
  const int l = lane & 15, hi = lane >> 4;
#pragma unroll
  for (int r = 0; r < 8; ++r)
    out[(size_t)(row_base + hi * 8 + r) * ld + col_base + l] = (_Float16)c[r];
}

// TDM: DMA one 16x256 f32 tile (rows contiguous, row stride 256) into LDS,
// inserting 8 DWORDs of padding after every 256 DWORDs so the LDS image has
// row stride 264 (bank stagger).  Issued by one wave; completion via TENSORcnt.
__device__ __forceinline__ void tdm_load_tile16x256(const float* gsrc,
                                                    void* lds_dst) {
  unsigned long long ga = (unsigned long long)(const void*)gsrc;
  // Flat LDS-aperture address: low 32 bits are the LDS byte offset.
  unsigned int lds_off = (unsigned int)(unsigned long long)lds_dst;
  u32x4 g0 = {};
  g0[0] = 1u;                                       // count=1 (valid, user)
  g0[1] = lds_off;                                  // lds_addr
  g0[2] = (unsigned int)(ga & 0xffffffffull);       // global_addr[31:0]
  g0[3] = (unsigned int)((ga >> 32) & 0x01ffffffull) | (2u << 30);  // +type=2
  i32x8 g1 = {};
  //        data_size=4B   pad_en      interval=256dw  amount=8dw
  g1[0] = (int)((2u << 16) | (1u << 20) | (7u << 22) | (7u << 25));
  g1[1] = (int)(256u << 16);   // tensor_dim0 = 256        (bits 79:48 lo)
  g1[2] = (int)(16u << 16);    // tensor_dim1 = 16         (bits 111:80 lo)
  g1[3] = (int)(256u << 16);   // tile_dim0   = 256        (bits 127:112)
  g1[4] = 16;                  // tile_dim1   = 16 (tile_dim2 = 0 = unused)
  g1[5] = 256;                 // tensor_dim0_stride = 256
  i32x4 gz = {};
#if __clang_major__ >= 23
  i32x8 g4 = {};
  __builtin_amdgcn_tensor_load_to_lds(g0, g1, gz, gz, g4, 0);
#else
  __builtin_amdgcn_tensor_load_to_lds(g0, g1, gz, gz, 0);
#endif
  __builtin_amdgcn_s_wait_tensorcnt(0);
}

// ---------------------------------------------------------------------------
__global__ void k_cvt(const float* __restrict__ in, _Float16* __restrict__ out,
                      int n) {
  int i = blockIdx.x * blockDim.x + threadIdx.x;
  if (i < n) out[i] = (_Float16)in[i];
}

__global__ void k_init_slots(const float* __restrict__ noise,
                             const float* __restrict__ mu,
                             const float* __restrict__ sigma,
                             float* __restrict__ slots) {
  int idx = blockIdx.x * blockDim.x + threadIdx.x;  // 32*16*256
  if (idx >= NB * NSLOTP * DIMD) return;
  int d = idx & 255, i = (idx >> 8) & 15, b = idx >> 12;
  float v = 0.f;
  if (i < 8) v = mu[d] + sigma[d] * noise[((size_t)b * 8 + i) * DIMD + d];
  slots[idx] = v;
}

// ---------------------------------------------------------------------------
// Stage 1: LN(inputs), k = x@wk.T+bk (row-major f16), v -> vT[d][j] f16.
// One block = 16 tokens, 8 waves, each wave 4 of 32 (output,tile) tasks.
__global__ __launch_bounds__(256) void k_ln_kv(
    const float* __restrict__ inputs, const float* __restrict__ lnw,
    const float* __restrict__ lnb, const _Float16* __restrict__ wk_h,
    const float* __restrict__ bk, const _Float16* __restrict__ wv_h,
    const float* __restrict__ bv, _Float16* __restrict__ kout,
    _Float16* __restrict__ vT) {
  __shared__ float sx[16][264];
  __shared__ _Float16 xh[16][264];
  __shared__ float red[2][16][16];
  __shared__ float ms[16][2];
  const int tid = threadIdx.x;
  const size_t tok0 = (size_t)blockIdx.x * 16;

  if (tid < 32) {  // wave 0: DMA the input tile into LDS (padded rows)
    tdm_load_tile16x256(inputs + tok0 * DIMD, (void*)&sx[0][0]);
  }
  __syncthreads();
  {
    int row = tid >> 4, part = tid & 15;
    float s = 0.f, s2 = 0.f;
#pragma unroll
    for (int i = 0; i < 16; ++i) {
      float v = sx[row][part + 16 * i];
      s += v; s2 += v * v;
    }
    red[0][row][part] = s; red[1][row][part] = s2;
  }
  __syncthreads();
  if (tid < 16) {
    float s = 0.f, s2 = 0.f;
    for (int i = 0; i < 16; ++i) { s += red[0][tid][i]; s2 += red[1][tid][i]; }
    float m = s * (1.f / 256.f);
    float var = s2 * (1.f / 256.f) - m * m;
    ms[tid][0] = m; ms[tid][1] = rsqrtf(var + 1e-5f);
  }
  __syncthreads();
  for (int idx = tid; idx < 16 * DIMD; idx += 256) {
    int row = idx >> 8, col = idx & 255;
    float v = (sx[row][col] - ms[row][0]) * ms[row][1] * lnw[col] + lnb[col];
    xh[row][col] = (_Float16)v;
  }
  __syncthreads();

  const int w = tid >> 5, lane = tid & 31, l = lane & 15, hi = lane >> 4;
  v16h a[8];
#pragma unroll
  for (int kk = 0; kk < 8; ++kk) a[kk] = fragA(&xh[0][0], 0, kk * 32, 264);

  for (int task = w; task < 32; task += 8) {
    const bool isv = task >= 16;
    const int tile = task & 15;
    const _Float16* W = isv ? wv_h : wk_h;
    const float* bias = isv ? bv : bk;
    float bsc = bias[tile * 16 + l];
    v8f c;
#pragma unroll
    for (int r = 0; r < 8; ++r) c[r] = bsc;
    v16h bcur = fragB(W, tile * 16, 0, DIMD);
#pragma unroll
    for (int kk = 0; kk < 7; ++kk) {            // load-next / compute-current
      v16h bnxt = fragB(W, tile * 16, (kk + 1) * 32, DIMD);
      c = wmma32(a[kk], bcur, c);
      bcur = bnxt;
    }
    c = wmma32(a[7], bcur, c);
    if (!isv) {
      stC_f16(kout + tok0 * DIMD, 0, tile * 16, DIMD, c);
    } else {
      const int bidx = (int)(tok0 >> 12);
      const int j0 = (int)(tok0 & (NTOK - 1));
      _Float16* vb = vT + (size_t)bidx * DIMD * NTOK;
      v8h h;
#pragma unroll
      for (int r = 0; r < 8; ++r) h[r] = (_Float16)c[r];
      *(v8h*)(vb + (size_t)(tile * 16 + l) * NTOK + j0 + hi * 8) = h;
    }
  }
}

// ---------------------------------------------------------------------------
// s_n = LN(slots); q = s_n @ wq.T + bq   (one block per batch, pad rows zero)
__global__ __launch_bounds__(256) void k_qproj(
    const float* __restrict__ slots, const float* __restrict__ lnw,
    const float* __restrict__ lnb, const _Float16* __restrict__ wq_h,
    const float* __restrict__ bq, _Float16* __restrict__ q) {
  __shared__ float sx[16][264];
  __shared__ _Float16 xh[16][264];
  __shared__ float red[2][16][16];
  __shared__ float ms[16][2];
  const int b = blockIdx.x, tid = threadIdx.x;
  const float* sb = slots + (size_t)b * NSLOTP * DIMD;

  for (int idx = tid; idx < 16 * DIMD; idx += 256) {
    int row = idx >> 8, col = idx & 255;
    sx[row][col] = (row < 8) ? sb[idx] : 0.f;
  }
  __syncthreads();
  {
    int row = tid >> 4, part = tid & 15;
    float s = 0.f, s2 = 0.f;
#pragma unroll
    for (int i = 0; i < 16; ++i) {
      float v = sx[row][part + 16 * i];
      s += v; s2 += v * v;
    }
    red[0][row][part] = s; red[1][row][part] = s2;
  }
  __syncthreads();
  if (tid < 16) {
    float s = 0.f, s2 = 0.f;
    for (int i = 0; i < 16; ++i) { s += red[0][tid][i]; s2 += red[1][tid][i]; }
    float m = s * (1.f / 256.f);
    float var = s2 * (1.f / 256.f) - m * m;
    ms[tid][0] = m; ms[tid][1] = rsqrtf(var + 1e-5f);
  }
  __syncthreads();
  for (int idx = tid; idx < 16 * DIMD; idx += 256) {
    int row = idx >> 8, col = idx & 255;
    float v = (row < 8)
        ? (sx[row][col] - ms[row][0]) * ms[row][1] * lnw[col] + lnb[col]
        : 0.f;
    xh[row][col] = (_Float16)v;
  }
  __syncthreads();

  const int w = tid >> 5, lane = tid & 31, l = lane & 15;
  v16h a[8];
#pragma unroll
  for (int kk = 0; kk < 8; ++kk) a[kk] = fragA(&xh[0][0], 0, kk * 32, 264);
  _Float16* qb = q + (size_t)b * NSLOTP * DIMD;
  for (int tile = w; tile < 16; tile += 8) {
    float bsc = bq[tile * 16 + l];
    v8f c;
#pragma unroll
    for (int r = 0; r < 8; ++r) c[r] = bsc;
    v16h bcur = fragB(wq_h, tile * 16, 0, DIMD);
#pragma unroll
    for (int kk = 0; kk < 7; ++kk) {
      v16h bnxt = fragB(wq_h, tile * 16, (kk + 1) * 32, DIMD);
      c = wmma32(a[kk], bcur, c);
      bcur = bnxt;
    }
    c = wmma32(a[7], bcur, c);
    stC_f16(qb, 0, tile * 16, DIMD, c);
  }
}

// ---------------------------------------------------------------------------
// dots = scale * q k^T; softmax over slots per token column; store attn f16
// (unnormalized over j; row-normalization folded into k_updates).
// grid = (8 token-chunks, 32 batches), 256 threads.
__global__ __launch_bounds__(256) void k_dots(const _Float16* __restrict__ q,
                                              const _Float16* __restrict__ kmat,
                                              _Float16* __restrict__ attn) {
  __shared__ float dots[16][520];
  const int chunk = blockIdx.x;
  const int b = blockIdx.y;
  const int tid = threadIdx.x;
  const int w = tid >> 5, lane = tid & 31, l = lane & 15;
  const _Float16* qb = q + (size_t)b * NSLOTP * DIMD;
  const _Float16* kb = kmat + (size_t)b * NTOK * DIMD;
  const int tok0 = chunk * 512;

  v16h a[8];
#pragma unroll
  for (int kk = 0; kk < 8; ++kk) a[kk] = fragA(qb, 0, kk * 32, DIMD);

  for (int t = w; t < 32; t += 8) {
    if (t + 8 < 32)  // prefetch next tile's k rows toward L2/L0
      __builtin_prefetch(kb + (size_t)(tok0 + (t + 8) * 16 + l) * DIMD, 0, 1);
    v8f c = {};
    v16h bcur = fragB(kb, tok0 + t * 16, 0, DIMD);
#pragma unroll
    for (int kk = 0; kk < 7; ++kk) {
      v16h bnxt = fragB(kb, tok0 + t * 16, (kk + 1) * 32, DIMD);
      c = wmma32(a[kk], bcur, c);
      bcur = bnxt;
    }
    c = wmma32(a[7], bcur, c);
#pragma unroll
    for (int r = 0; r < 8; ++r) c[r] *= 0.0625f;  // d^-0.5
    stC_f32(&dots[0][0], 0, t * 16, 520, c);
  }
  __syncthreads();

  _Float16* ab = attn + (size_t)b * NSLOTP * NTOK + tok0;
  for (int col = tid; col < 512; col += 256) {
    float m = -1e30f;
#pragma unroll
    for (int i = 0; i < 8; ++i) m = fmaxf(m, dots[i][col]);
    float e[8], s = 0.f;
#pragma unroll
    for (int i = 0; i < 8; ++i) { e[i] = __expf(dots[i][col] - m); s += e[i]; }
    float inv = 1.f / s;
#pragma unroll
    for (int i = 0; i < 8; ++i)
      ab[(size_t)i * NTOK + col] = (_Float16)(e[i] * inv + 1e-8f);
#pragma unroll
    for (int i = 8; i < 16; ++i) ab[(size_t)i * NTOK + col] = (_Float16)0.f;
  }
}

// Per-slot inverse row sums (deterministic: no atomics).
__global__ __launch_bounds__(256) void k_rowsum(const _Float16* __restrict__ attn,
                                                float* __restrict__ invsum) {
  __shared__ float red[8][33];
  const int b = blockIdx.x, tid = threadIdx.x;
  const int row = tid >> 5, lane = tid & 31;
  const _Float16* ab = attn + (size_t)b * NSLOTP * NTOK + (size_t)row * NTOK;
  float s = 0.f;
  for (int i = lane; i < NTOK; i += 32) s += (float)ab[i];
  red[row][lane] = s;
  __syncthreads();
  if (tid < 16) {
    if (tid < 8) {
      float t = 0.f;
      for (int i = 0; i < 32; ++i) t += red[tid][i];
      invsum[b * 16 + tid] = 1.f / t;
    } else {
      invsum[b * 16 + tid] = 0.f;  // zeroes the padded update rows
    }
  }
}

// updates = (attn / rowsum) @ v  via  attn @ vT with post-accumulate rescale.
// grid = 32 batches, 512 threads = 16 waves, wave = one 16-wide d tile.
// Double-buffered over the K=4096 contraction.
__global__ __launch_bounds__(512) void k_updates(
    const _Float16* __restrict__ attn, const _Float16* __restrict__ vT,
    const float* __restrict__ invsum, float* __restrict__ updates) {
  const int b = blockIdx.x;
  const int w = threadIdx.x >> 5;
  const int lane = threadIdx.x & 31, l = lane & 15, hi = lane >> 4;
  const _Float16* ab = attn + (size_t)b * NSLOTP * NTOK;
  const _Float16* vb = vT + (size_t)b * DIMD * NTOK;
  v8f c = {};
  v16h acur = fragA(ab, 0, 0, NTOK);
  v16h bcur = fragB(vb, w * 16, 0, NTOK);
#pragma unroll 4
  for (int kk = 0; kk < NTOK / 32 - 1; ++kk) {
    if (kk + 8 < NTOK / 32)  // stream-ahead prefetch of vT
      __builtin_prefetch(vb + (size_t)(w * 16 + l) * NTOK + (kk + 8) * 32, 0, 1);
    v16h anxt = fragA(ab, 0, (kk + 1) * 32, NTOK);
    v16h bnxt = fragB(vb, w * 16, (kk + 1) * 32, NTOK);
    c = wmma32(acur, bcur, c);
    acur = anxt; bcur = bnxt;
  }
  c = wmma32(acur, bcur, c);
  float* ub = updates + (size_t)b * NSLOTP * DIMD;
#pragma unroll
  for (int r = 0; r < 8; ++r) {
    float sc = invsum[b * 16 + hi * 8 + r];
    ub[(size_t)(hi * 8 + r) * DIMD + w * 16 + l] = c[r] * sc;
  }
}

// ---------------------------------------------------------------------------
// GRU cell + LN + residual MLP.  One block per batch; gi/gh staged in ws.
__global__ __launch_bounds__(256) void k_gru(
    const float* __restrict__ updates, float* __restrict__ slots,
    float* __restrict__ gi_g, float* __restrict__ gh_g,
    const _Float16* __restrict__ wih_h, const float* __restrict__ bih,
    const _Float16* __restrict__ whh_h, const float* __restrict__ bhh,
    const _Float16* __restrict__ fc1_h, const float* __restrict__ fc1b,
    const _Float16* __restrict__ fc2_h, const float* __restrict__ fc2b,
    const float* __restrict__ lnw, const float* __restrict__ lnb) {
  __shared__ _Float16 AH0[16][264];
  __shared__ _Float16 AH1[16][264];
  __shared__ float SP[16][264];
  __shared__ float red[2][16][16];
  __shared__ float ms[16][2];
  const int b = blockIdx.x, tid = threadIdx.x;
  const float* ub = updates + (size_t)b * NSLOTP * DIMD;
  float* sb = slots + (size_t)b * NSLOTP * DIMD;
  float* GI = gi_g + (size_t)b * NSLOTP * 768;
  float* GH = gh_g + (size_t)b * NSLOTP * 768;

  for (int idx = tid; idx < 16 * DIMD; idx += 256) {
    int row = idx >> 8, col = idx & 255;
    float u = ub[idx], s = sb[idx];
    AH0[row][col] = (_Float16)u;
    AH1[row][col] = (_Float16)s;
    SP[row][col] = s;
  }
  __syncthreads();

  {  // gi (waves 0-3) / gh (waves 4-7): [16,256] @ [256,768]^T
    const int w = tid >> 5, half = w >> 2, wsub = w & 3;
    const _Float16* A = half ? &AH1[0][0] : &AH0[0][0];
    const _Float16* W = half ? whh_h : wih_h;
    const float* bias = half ? bhh : bih;
    float* G = half ? GH : GI;
    const int lane = tid & 31, l = lane & 15;
    v16h a[8];
#pragma unroll
    for (int kk = 0; kk < 8; ++kk) a[kk] = fragA(A, 0, kk * 32, 264);
    for (int t = wsub; t < 48; t += 4) {
      float bsc = bias[t * 16 + l];
      v8f c;
#pragma unroll
      for (int r = 0; r < 8; ++r) c[r] = bsc;
      v16h bcur = fragB(W, t * 16, 0, DIMD);
#pragma unroll
      for (int kk = 0; kk < 7; ++kk) {
        v16h bnxt = fragB(W, t * 16, (kk + 1) * 32, DIMD);
        c = wmma32(a[kk], bcur, c);
        bcur = bnxt;
      }
      c = wmma32(a[7], bcur, c);
      stC_f32(G, 0, t * 16, 768, c);
    }
  }
  __threadfence_block();
  __syncthreads();

  for (int idx = tid; idx < 16 * DIMD; idx += 256) {  // gates
    int row = idx >> 8, col = idx & 255;
    float ir = GI[row * 768 + col], iz = GI[row * 768 + 256 + col],
          in_ = GI[row * 768 + 512 + col];
    float hr = GH[row * 768 + col], hz = GH[row * 768 + 256 + col],
          hn = GH[row * 768 + 512 + col];
    float r = 1.f / (1.f + __expf(-(ir + hr)));
    float z = 1.f / (1.f + __expf(-(iz + hz)));
    float nn = tanhf(in_ + r * hn);
    SP[row][col] = (1.f - z) * nn + z * SP[row][col];
  }
  __syncthreads();

  {  // LN stats of gru output
    int row = tid >> 4, part = tid & 15;
    float s = 0.f, s2 = 0.f;
#pragma unroll
    for (int i = 0; i < 16; ++i) {
      float v = SP[row][part + 16 * i];
      s += v; s2 += v * v;
    }
    red[0][row][part] = s; red[1][row][part] = s2;
  }
  __syncthreads();
  if (tid < 16) {
    float s = 0.f, s2 = 0.f;
    for (int i = 0; i < 16; ++i) { s += red[0][tid][i]; s2 += red[1][tid][i]; }
    float m = s * (1.f / 256.f);
    float var = s2 * (1.f / 256.f) - m * m;
    ms[tid][0] = m; ms[tid][1] = rsqrtf(var + 1e-5f);
  }
  __syncthreads();
  for (int idx = tid; idx < 16 * DIMD; idx += 256) {
    int row = idx >> 8, col = idx & 255;
    float v = (SP[row][col] - ms[row][0]) * ms[row][1] * lnw[col] + lnb[col];
    AH0[row][col] = (_Float16)v;
  }
  __syncthreads();

  const int w = tid >> 5, lane = tid & 31, l = lane & 15, hi = lane >> 4;
  {  // t1 = relu(ff @ fc1.T + b)
    v16h a[8];
#pragma unroll
    for (int kk = 0; kk < 8; ++kk) a[kk] = fragA(&AH0[0][0], 0, kk * 32, 264);
    for (int t = w; t < 16; t += 8) {
      float bsc = fc1b[t * 16 + l];
      v8f c;
#pragma unroll
      for (int r = 0; r < 8; ++r) c[r] = bsc;
      v16h bcur = fragB(fc1_h, t * 16, 0, DIMD);
#pragma unroll
      for (int kk = 0; kk < 7; ++kk) {
        v16h bnxt = fragB(fc1_h, t * 16, (kk + 1) * 32, DIMD);
        c = wmma32(a[kk], bcur, c);
        bcur = bnxt;
      }
      c = wmma32(a[7], bcur, c);
#pragma unroll
      for (int r = 0; r < 8; ++r)
        AH1[hi * 8 + r][t * 16 + l] = (_Float16)fmaxf(c[r], 0.f);
    }
  }
  __syncthreads();
  {  // slots = gru_out + t1 @ fc2.T + b
    v16h a[8];
#pragma unroll
    for (int kk = 0; kk < 8; ++kk) a[kk] = fragA(&AH1[0][0], 0, kk * 32, 264);
    for (int t = w; t < 16; t += 8) {
      float bsc = fc2b[t * 16 + l];
      v8f c;
#pragma unroll
      for (int r = 0; r < 8; ++r) c[r] = bsc;
      v16h bcur = fragB(fc2_h, t * 16, 0, DIMD);
#pragma unroll
      for (int kk = 0; kk < 7; ++kk) {
        v16h bnxt = fragB(fc2_h, t * 16, (kk + 1) * 32, DIMD);
        c = wmma32(a[kk], bcur, c);
        bcur = bnxt;
      }
      c = wmma32(a[7], bcur, c);
#pragma unroll
      for (int r = 0; r < 8; ++r) {
        int rr = hi * 8 + r, cc = t * 16 + l;
        sb[(size_t)rr * DIMD + cc] = SP[rr][cc] + c[r];
      }
    }
  }
}

__global__ void k_copy_out(const float* __restrict__ slots,
                           float* __restrict__ out) {
  int idx = blockIdx.x * blockDim.x + threadIdx.x;  // 32*8*256
  if (idx >= NB * 8 * DIMD) return;
  int d = idx & 255, i = (idx >> 8) & 7, b = idx >> 11;
  out[idx] = slots[((size_t)b * NSLOTP + i) * DIMD + d];
}

// ---------------------------------------------------------------------------
extern "C" void kernel_launch(void* const* d_in, const int* in_sizes, int n_in,
                              void* d_out, int out_size, void* d_ws,
                              size_t ws_size, hipStream_t stream) {
  (void)in_sizes; (void)n_in; (void)out_size; (void)ws_size;
  const float* inputs = (const float*)d_in[0];
  const float* noise  = (const float*)d_in[1];
  const float* mu     = (const float*)d_in[2];
  const float* sigma  = (const float*)d_in[3];
  const float* wq  = (const float*)d_in[4];  const float* bq  = (const float*)d_in[5];
  const float* wk  = (const float*)d_in[6];  const float* bk  = (const float*)d_in[7];
  const float* wv  = (const float*)d_in[8];  const float* bv  = (const float*)d_in[9];
  const float* wih = (const float*)d_in[10]; const float* bih = (const float*)d_in[11];
  const float* whh = (const float*)d_in[12]; const float* bhh = (const float*)d_in[13];
  const float* fc1w = (const float*)d_in[14]; const float* fc1b = (const float*)d_in[15];
  const float* fc2w = (const float*)d_in[16]; const float* fc2b = (const float*)d_in[17];
  const float* lin_w = (const float*)d_in[18]; const float* lin_b = (const float*)d_in[19];
  const float* lns_w = (const float*)d_in[20]; const float* lns_b = (const float*)d_in[21];
  const float* lnf_w = (const float*)d_in[22]; const float* lnf_b = (const float*)d_in[23];

  char* p = (char*)d_ws;
  auto take = [&](size_t bytes) -> char* {
    char* r = p;
    p += (bytes + 255) & ~(size_t)255;
    return r;
  };
  _Float16* wq_h  = (_Float16*)take((size_t)DIMD * DIMD * 2);
  _Float16* wk_h  = (_Float16*)take((size_t)DIMD * DIMD * 2);
  _Float16* wv_h  = (_Float16*)take((size_t)DIMD * DIMD * 2);
  _Float16* wih_h = (_Float16*)take((size_t)3 * DIMD * DIMD * 2);
  _Float16* whh_h = (_Float16*)take((size_t)3 * DIMD * DIMD * 2);
  _Float16* fc1_h = (_Float16*)take((size_t)HID * DIMD * 2);
  _Float16* fc2_h = (_Float16*)take((size_t)DIMD * HID * 2);
  _Float16* k_h   = (_Float16*)take((size_t)NB * NTOK * DIMD * 2);
  _Float16* vT_h  = (_Float16*)take((size_t)NB * DIMD * NTOK * 2);
  _Float16* q_h   = (_Float16*)take((size_t)NB * NSLOTP * DIMD * 2);
  _Float16* attn_h = (_Float16*)take((size_t)NB * NSLOTP * NTOK * 2);
  float* invsum  = (float*)take((size_t)NB * NSLOTP * 4);
  float* updates = (float*)take((size_t)NB * NSLOTP * DIMD * 4);
  float* slots   = (float*)take((size_t)NB * NSLOTP * DIMD * 4);
  float* gi_g    = (float*)take((size_t)NB * NSLOTP * 768 * 4);
  float* gh_g    = (float*)take((size_t)NB * NSLOTP * 768 * 4);

  const int TPB = 256;
  k_cvt<<<(DIMD * DIMD + TPB - 1) / TPB, TPB, 0, stream>>>(wq, wq_h, DIMD * DIMD);
  k_cvt<<<(DIMD * DIMD + TPB - 1) / TPB, TPB, 0, stream>>>(wk, wk_h, DIMD * DIMD);
  k_cvt<<<(DIMD * DIMD + TPB - 1) / TPB, TPB, 0, stream>>>(wv, wv_h, DIMD * DIMD);
  k_cvt<<<(3 * DIMD * DIMD + TPB - 1) / TPB, TPB, 0, stream>>>(wih, wih_h, 3 * DIMD * DIMD);
  k_cvt<<<(3 * DIMD * DIMD + TPB - 1) / TPB, TPB, 0, stream>>>(whh, whh_h, 3 * DIMD * DIMD);
  k_cvt<<<(HID * DIMD + TPB - 1) / TPB, TPB, 0, stream>>>(fc1w, fc1_h, HID * DIMD);
  k_cvt<<<(DIMD * HID + TPB - 1) / TPB, TPB, 0, stream>>>(fc2w, fc2_h, DIMD * HID);

  k_init_slots<<<(NB * NSLOTP * DIMD + TPB - 1) / TPB, TPB, 0, stream>>>(
      noise, mu, sigma, slots);

  k_ln_kv<<<NB * NTOK / 16, 256, 0, stream>>>(inputs, lin_w, lin_b, wk_h, bk,
                                              wv_h, bv, k_h, vT_h);

  for (int it = 0; it < 3; ++it) {
    k_qproj<<<NB, 256, 0, stream>>>(slots, lns_w, lns_b, wq_h, bq, q_h);
    k_dots<<<dim3(NTOK / 512, NB), 256, 0, stream>>>(q_h, k_h, attn_h);
    k_rowsum<<<NB, 256, 0, stream>>>(attn_h, invsum);
    k_updates<<<NB, 512, 0, stream>>>(attn_h, vT_h, invsum, updates);
    k_gru<<<NB, 256, 0, stream>>>(updates, slots, gi_g, gh_g, wih_h, bih,
                                  whh_h, bhh, fc1_h, fc1b, fc2_h, fc2b,
                                  lnf_w, lnf_b);
  }

  k_copy_out<<<(NB * 8 * DIMD + TPB - 1) / TPB, TPB, 0, stream>>>(
      slots, (float*)d_out);
}
